// SelfAttention_69380901700249
// MI455X (gfx1250) — matrix operand order repaired and checked by reading, not compiled, not run on hardware
//
#include <hip/hip_runtime.h>

// ---------------------------------------------------------------------------
// Self-attention forward, MI455X (gfx1250, wave32, WMMA, async global->LDS).
// fp32 inputs -> bf16 operands, f32 WMMA accumulation.
// Workspace (bf16): qb [N][H][L][D] | kb [N][H][L][D] | vb [N][H][D][L] |
//                   ob [N][L][H][D].  Needs 64 MB of d_ws.
// ---------------------------------------------------------------------------

#define L_SEQ 2048
#define NBATCH 4
#define EMB 1024
#define NHEAD 16
#define HDIM 64
#define NCHUNK (L_SEQ / 64)

typedef __bf16 bf16;
typedef __attribute__((ext_vector_type(16))) __bf16 v16bf;
typedef __attribute__((ext_vector_type(8)))  float  v8f;
typedef __attribute__((ext_vector_type(4)))  unsigned int u32x4;
typedef __attribute__((ext_vector_type(4)))  int i32x4;
typedef __attribute__((ext_vector_type(4)))  float f32x4;

union Frag { v16bf v; u32x4 q[2]; };

#if defined(__AMDGCN__) && \
    __has_builtin(__builtin_amdgcn_global_load_async_to_lds_b128) && \
    __has_builtin(__builtin_amdgcn_s_wait_asynccnt)
#define USE_ASYNC 1
#else
#define USE_ASYNC 0
#endif

#if USE_ASYNC
// builtin signature: (v4i addrspace(1)* src, v4i addrspace(3)* dst, imm off, imm cpol)
typedef __attribute__((address_space(1))) i32x4* gp128_t;
typedef __attribute__((address_space(3))) i32x4* lp128_t;
__device__ inline void async_b128(const void* g, void* l) {
  // global_load_async_to_lds_b128: per-lane 16B copy, tracked by ASYNCcnt
  __builtin_amdgcn_global_load_async_to_lds_b128((gp128_t)g, (lp128_t)l, 0, 0);
}
#endif

__device__ inline void wait_async0() {
#if USE_ASYNC
  __builtin_amdgcn_s_wait_asynccnt(0);
#endif
}

// copy 32 contiguous bf16 (64B): async (4x b128) or sync fallback
__device__ inline void copy64B(const bf16* g, bf16* l) {
#if USE_ASYNC
  async_b128(g, l);           async_b128(g + 8, l + 8);
  async_b128(g + 16, l + 16); async_b128(g + 24, l + 24);
#else
  u32x4* d = reinterpret_cast<u32x4*>(l);
  const u32x4* s = reinterpret_cast<const u32x4*>(g);
  d[0] = s[0]; d[1] = s[1]; d[2] = s[2]; d[3] = s[3];
#endif
}

__device__ inline v8f zero_v8f() {
  v8f z;
#pragma unroll
  for (int i = 0; i < 8; ++i) z[i] = 0.f;
  return z;
}

__device__ inline v8f wmma_bf16(v16bf a, v16bf b, v8f c) {
  // D(f32 16x16) = A(bf16 16x32) * B(bf16 32x16) + C
  return __builtin_amdgcn_wmma_f32_16x16x32_bf16(
      false, a, false, b, (short)0, c, false, false);
}

// ---------------------------------------------------------------------------
// Kernel 1: fused Q/K/V projections.  grid = (16 ntiles, 64 mtiles, 3 qkv)
// Q,K stored [N][H][L][D]; V stored transposed [N][H][D][L].
// ---------------------------------------------------------------------------
__global__ __launch_bounds__(256) void proj_qkv(
    const float* __restrict__ query, const float* __restrict__ keys,
    const float* __restrict__ values,
    const float* __restrict__ Wq, const float* __restrict__ Wk,
    const float* __restrict__ Wv,
    const float* __restrict__ bq, const float* __restrict__ bk,
    const float* __restrict__ bv,
    unsigned short* __restrict__ qb, unsigned short* __restrict__ kb,
    unsigned short* __restrict__ vb)
{
  __shared__ __align__(16) bf16 As[128 * 32];  // [m][k]
  __shared__ __align__(16) bf16 Ws[64 * 32];   // [n][k] (transposed)

  const int z = blockIdx.z;
  const float* A    = (z == 0) ? query : (z == 1) ? keys : values;
  const float* W    = (z == 0) ? Wq    : (z == 1) ? Wk   : Wv;
  const float* bias = (z == 0) ? bq    : (z == 1) ? bk   : bv;
  bf16* out = reinterpret_cast<bf16*>((z == 0) ? qb : (z == 1) ? kb : vb);

  const int t = threadIdx.x;
  const int w = t >> 5, lane = t & 31;
  const int ln = lane & 15, kh = lane >> 4;
  const int m0 = blockIdx.y * 128;
  const int n0 = blockIdx.x * 64;

  v8f acc[4];
#pragma unroll
  for (int j = 0; j < 4; ++j) acc[j] = zero_v8f();

  const int arow = t >> 1, ahalf = t & 1;   // A staging: 128 rows x 2 half-rows
  const int wk = t >> 3, wn = (t & 7) * 8;  // W staging: 32 k-rows x 8 n-octets

  for (int kk = 0; kk < EMB; kk += 32) {
    {  // stage A tile, fp32 -> bf16
      const float* src = A + (size_t)(m0 + arow) * EMB + kk + ahalf * 16;
      bf16* dst = &As[arow * 32 + ahalf * 16];
#pragma unroll
      for (int i4 = 0; i4 < 4; ++i4) {
        f32x4 v = *reinterpret_cast<const f32x4*>(src + i4 * 4);
        dst[i4 * 4 + 0] = (bf16)v[0]; dst[i4 * 4 + 1] = (bf16)v[1];
        dst[i4 * 4 + 2] = (bf16)v[2]; dst[i4 * 4 + 3] = (bf16)v[3];
      }
    }
    {  // stage W tile transposed: Ws[n][k]
      const float* src = W + (size_t)(kk + wk) * EMB + n0 + wn;
#pragma unroll
      for (int i4 = 0; i4 < 2; ++i4) {
        f32x4 v = *reinterpret_cast<const f32x4*>(src + i4 * 4);
#pragma unroll
        for (int e = 0; e < 4; ++e)
          Ws[(wn + i4 * 4 + e) * 32 + wk] = (bf16)v[e];
      }
    }
    if (kk + 32 < EMB) {  // prefetch next tiles (global_prefetch_b8)
      __builtin_prefetch(A + (size_t)(m0 + arow) * EMB + kk + 32 + ahalf * 16, 0, 0);
      __builtin_prefetch(W + (size_t)(kk + 32 + wk) * EMB + n0 + wn, 0, 0);
    }
    __syncthreads();

    Frag a;  // A-frag: lane row = ln, runs at kh*8 and 16+kh*8
    a.q[0] = *reinterpret_cast<const u32x4*>(&As[(w * 16 + ln) * 32 + kh * 8]);
    a.q[1] = *reinterpret_cast<const u32x4*>(&As[(w * 16 + ln) * 32 + 16 + kh * 8]);
#pragma unroll
    for (int j = 0; j < 4; ++j) {
      Frag b;  // B-frag: lane col = ln, runs at kh*8 and 16+kh*8
      b.q[0] = *reinterpret_cast<const u32x4*>(&Ws[(j * 16 + ln) * 32 + kh * 8]);
      b.q[1] = *reinterpret_cast<const u32x4*>(&Ws[(j * 16 + ln) * 32 + 16 + kh * 8]);
      acc[j] = wmma_bf16(a.v, b.v, acc[j]);
    }
    __syncthreads();
  }

#pragma unroll
  for (int j = 0; j < 4; ++j) {
    const int e = n0 + j * 16 + ln;
    const float bval = bias[e];
    const int h = e >> 6, d = e & 63;
#pragma unroll
    for (int r = 0; r < 8; ++r) {
      const int m = m0 + w * 16 + r + 8 * kh;
      const int nb = m >> 11, l = m & (L_SEQ - 1);
      const size_t idx = (z == 2)
          ? (((size_t)(nb * NHEAD + h) * HDIM + d) * L_SEQ + l)   // V: [N][H][D][L]
          : (((size_t)(nb * NHEAD + h) * L_SEQ + l) * HDIM + d);  // Q,K: [N][H][L][D]
      out[idx] = (bf16)(acc[j][r] + bval);
    }
  }
}

// ---------------------------------------------------------------------------
// Kernel 2: flash attention with async double-buffered K/V staging.
// grid = (L/64 qblocks, N*H), block 128 (4 waves); wave: 16 q-rows x D=64.
// ---------------------------------------------------------------------------
__global__ __launch_bounds__(128) void attn_flash(
    const unsigned short* __restrict__ qbu, const unsigned short* __restrict__ kbu,
    const unsigned short* __restrict__ vbu, const int* __restrict__ mask,
    unsigned short* __restrict__ obu)
{
  __shared__ __align__(16) bf16 Qs[64 * 64];         // [qrow][d]
  __shared__ __align__(16) bf16 Ks[2][64 * 64];      // [kpos][d], double buffered
  __shared__ __align__(16) bf16 Vs[2][64 * 64];      // [d][kpos], double buffered
  __shared__ __align__(16) bf16 Ps[4 * 16 * 64];     // per-wave P tile [qrow][kpos]

  const bf16* qb = reinterpret_cast<const bf16*>(qbu);
  const bf16* kb = reinterpret_cast<const bf16*>(kbu);
  const bf16* vb = reinterpret_cast<const bf16*>(vbu);
  bf16* ob = reinterpret_cast<bf16*>(obu);

  const int t = threadIdx.x;
  const int w = t >> 5, lane = t & 31;
  const int ln = lane & 15, kh = lane >> 4;
  const int q0 = blockIdx.x * 64;
  const int nb = blockIdx.y >> 4, h = blockIdx.y & 15;
  const size_t headbase = (size_t)(nb * NHEAD + h) * L_SEQ * HDIM;

  const bf16* kbase = kb + headbase;  // [L][D]
  const bf16* vbase = vb + headbase;  // [D][L]
  const int vrow = t >> 1, vhalf = t & 1;

  // stage one 64-key chunk (K rows + V rows) into buffer `buf`
  auto stage_chunk = [&](int kpos0, int buf) {
    copy64B(kbase + (size_t)kpos0 * HDIM + t * 32, &Ks[buf][t * 32]);
    copy64B(vbase + (size_t)vrow * L_SEQ + kpos0 + vhalf * 32,
            &Vs[buf][vrow * 64 + vhalf * 32]);
  };

  // prologue: Q block + chunk 0
  copy64B(qb + headbase + (size_t)q0 * HDIM + t * 32, &Qs[t * 32]);
  stage_chunk(0, 0);
  wait_async0();
  __syncthreads();

  Frag qf[2];  // Q A-fragments, held in registers for the whole kernel
#pragma unroll
  for (int kd = 0; kd < 2; ++kd) {
    qf[kd].q[0] = *reinterpret_cast<const u32x4*>(&Qs[(w * 16 + ln) * 64 + kd * 32 + kh * 8]);
    qf[kd].q[1] = *reinterpret_cast<const u32x4*>(&Qs[(w * 16 + ln) * 64 + kd * 32 + 16 + kh * 8]);
  }

  float mrun[8], lrun[8];
#pragma unroll
  for (int r = 0; r < 8; ++r) { mrun[r] = -3.0e38f; lrun[r] = 0.f; }
  v8f o[4];
#pragma unroll
  for (int j = 0; j < 4; ++j) o[j] = zero_v8f();

  for (int kc = 0; kc < NCHUNK; ++kc) {
    const int cur = kc & 1;
    const int kpos0 = kc * 64;
    if (kc + 1 < NCHUNK) stage_chunk((kc + 1) * 64, cur ^ 1);  // async prefetch

    // S = Q * K^T  (16x64 per wave, f32 accum)
    v8f s[4];
#pragma unroll
    for (int j = 0; j < 4; ++j) s[j] = zero_v8f();
#pragma unroll
    for (int j = 0; j < 4; ++j)
#pragma unroll
      for (int kd = 0; kd < 2; ++kd) {
        Frag b;  // B-frag: col = key pos, d-runs at kd*32 + kh*8 and kd*32 + 16 + kh*8
        b.q[0] = *reinterpret_cast<const u32x4*>(&Ks[cur][(j * 16 + ln) * 64 + kd * 32 + kh * 8]);
        b.q[1] = *reinterpret_cast<const u32x4*>(&Ks[cur][(j * 16 + ln) * 64 + kd * 32 + 16 + kh * 8]);
        s[j] = wmma_bf16(qf[kd].v, b.v, s[j]);
      }

    // scale by 1/sqrt(D) and apply key mask
#pragma unroll
    for (int j = 0; j < 4; ++j) {
      const int kp = kpos0 + j * 16 + ln;
      const int mv = mask[nb * L_SEQ + kp];
#pragma unroll
      for (int r = 0; r < 8; ++r) {
        const float sv = s[j][r] * 0.125f;
        s[j][r] = (mv == 0) ? -1.0e20f : sv;
      }
    }

    // online softmax: each row lives across 16 lanes of a half-wave
    float fac[8];
#pragma unroll
    for (int r = 0; r < 8; ++r) {
      float mx = fmaxf(fmaxf(s[0][r], s[1][r]), fmaxf(s[2][r], s[3][r]));
#pragma unroll
      for (int off = 1; off < 16; off <<= 1) mx = fmaxf(mx, __shfl_xor(mx, off, 32));
      const float mnew = fmaxf(mrun[r], mx);
      fac[r] = __expf(mrun[r] - mnew);
      mrun[r] = mnew;
    }
    float rs[8];
#pragma unroll
    for (int r = 0; r < 8; ++r) rs[r] = 0.f;
#pragma unroll
    for (int j = 0; j < 4; ++j)
#pragma unroll
      for (int r = 0; r < 8; ++r) {
        const float p = __expf(s[j][r] - mrun[r]);
        s[j][r] = p;
        rs[r] += p;
      }
#pragma unroll
    for (int r = 0; r < 8; ++r) {
#pragma unroll
      for (int off = 1; off < 16; off <<= 1) rs[r] += __shfl_xor(rs[r], off, 32);
      lrun[r] = lrun[r] * fac[r] + rs[r];
    }
#pragma unroll
    for (int j = 0; j < 4; ++j)
#pragma unroll
      for (int r = 0; r < 8; ++r) o[j][r] *= fac[r];

    // P (C-layout) -> LDS -> A-fragment layout (per-wave region, same-wave order)
#pragma unroll
    for (int j = 0; j < 4; ++j)
#pragma unroll
      for (int r = 0; r < 8; ++r)
        Ps[(w * 16 + r + 8 * kh) * 64 + j * 16 + ln] = (bf16)s[j][r];

    Frag af[2];
#pragma unroll
    for (int kp = 0; kp < 2; ++kp) {
      af[kp].q[0] = *reinterpret_cast<const u32x4*>(&Ps[(w * 16 + ln) * 64 + kp * 32 + kh * 8]);
      af[kp].q[1] = *reinterpret_cast<const u32x4*>(&Ps[(w * 16 + ln) * 64 + kp * 32 + 16 + kh * 8]);
    }
    // O += P * V
#pragma unroll
    for (int dj = 0; dj < 4; ++dj)
#pragma unroll
      for (int kp = 0; kp < 2; ++kp) {
        Frag b;
        b.q[0] = *reinterpret_cast<const u32x4*>(&Vs[cur][(dj * 16 + ln) * 64 + kp * 32 + kh * 8]);
        b.q[1] = *reinterpret_cast<const u32x4*>(&Vs[cur][(dj * 16 + ln) * 64 + kp * 32 + 16 + kh * 8]);
        o[dj] = wmma_bf16(af[kp].v, b.v, o[dj]);
      }

    wait_async0();   // next chunk's async copies complete (own wave's share)
    __syncthreads(); // all waves done reading cur + all staging visible
  }

  // epilogue -> ob [N][L][H][D] ( == [N*L][E] row major, e = h*64+d )
#pragma unroll
  for (int j = 0; j < 4; ++j) {
    const int d = j * 16 + ln;
#pragma unroll
    for (int r = 0; r < 8; ++r) {
      const int l = q0 + w * 16 + r + 8 * kh;
      const float val = o[j][r] / lrun[r];
      ob[(((size_t)nb * L_SEQ + l) * NHEAD + h) * HDIM + d] = (bf16)val;
    }
  }
}

// ---------------------------------------------------------------------------
// Kernel 3: output projection. grid = (16 ntiles, 64 mtiles), block 256.
// A tile staged via async global->LDS (already bf16).
// ---------------------------------------------------------------------------
__global__ __launch_bounds__(256) void out_proj(
    const unsigned short* __restrict__ obu, const float* __restrict__ Wo,
    const float* __restrict__ bo, float* __restrict__ out)
{
  __shared__ __align__(16) bf16 As[128 * 32];
  __shared__ __align__(16) bf16 Ws[64 * 32];
  const bf16* A = reinterpret_cast<const bf16*>(obu);

  const int t = threadIdx.x;
  const int w = t >> 5, lane = t & 31;
  const int ln = lane & 15, kh = lane >> 4;
  const int m0 = blockIdx.y * 128, n0 = blockIdx.x * 64;

  v8f acc[4];
#pragma unroll
  for (int j = 0; j < 4; ++j) acc[j] = zero_v8f();

  const int arow = t >> 1, ahalf = t & 1;
  const int wk = t >> 3, wn = (t & 7) * 8;

  for (int kk = 0; kk < EMB; kk += 32) {
    {  // A tile: raw bf16 32B copy per thread (async if available)
      const bf16* src = A + (size_t)(m0 + arow) * EMB + kk + ahalf * 16;
      bf16* dst = &As[arow * 32 + ahalf * 16];
#if USE_ASYNC
      async_b128(src, dst); async_b128(src + 8, dst + 8);
#else
      u32x4* d4 = reinterpret_cast<u32x4*>(dst);
      d4[0] = reinterpret_cast<const u32x4*>(src)[0];
      d4[1] = reinterpret_cast<const u32x4*>(src)[1];
#endif
    }
    {  // Wo fp32 -> bf16, transposed
      const float* src = Wo + (size_t)(kk + wk) * EMB + n0 + wn;
#pragma unroll
      for (int i4 = 0; i4 < 2; ++i4) {
        f32x4 v = *reinterpret_cast<const f32x4*>(src + i4 * 4);
#pragma unroll
        for (int e = 0; e < 4; ++e)
          Ws[(wn + i4 * 4 + e) * 32 + wk] = (bf16)v[e];
      }
    }
    if (kk + 32 < EMB) {
      __builtin_prefetch(A + (size_t)(m0 + arow) * EMB + kk + 32 + ahalf * 16, 0, 0);
      __builtin_prefetch(Wo + (size_t)(kk + 32 + wk) * EMB + n0 + wn, 0, 0);
    }
    wait_async0();
    __syncthreads();

    Frag a;
    a.q[0] = *reinterpret_cast<const u32x4*>(&As[(w * 16 + ln) * 32 + kh * 8]);
    a.q[1] = *reinterpret_cast<const u32x4*>(&As[(w * 16 + ln) * 32 + 16 + kh * 8]);
#pragma unroll
    for (int j = 0; j < 4; ++j) {
      Frag b;
      b.q[0] = *reinterpret_cast<const u32x4*>(&Ws[(j * 16 + ln) * 32 + kh * 8]);
      b.q[1] = *reinterpret_cast<const u32x4*>(&Ws[(j * 16 + ln) * 32 + 16 + kh * 8]);
      acc[j] = wmma_bf16(a.v, b.v, acc[j]);
    }
    __syncthreads();
  }

#pragma unroll
  for (int j = 0; j < 4; ++j) {
    const int e = n0 + j * 16 + ln;
    const float bval = bo[e];
#pragma unroll
    for (int r = 0; r < 8; ++r) {
      const int m = m0 + w * 16 + r + 8 * kh;
      out[(size_t)m * EMB + e] = acc[j][r] + bval;
    }
  }
}

// ---------------------------------------------------------------------------
extern "C" void kernel_launch(void* const* d_in, const int* in_sizes, int n_in,
                              void* d_out, int out_size, void* d_ws, size_t ws_size,
                              hipStream_t stream) {
  const float* values = (const float*)d_in[0];
  const float* keys   = (const float*)d_in[1];
  const float* query  = (const float*)d_in[2];
  const int*   mask   = (const int*)d_in[3];
  const float* Wv = (const float*)d_in[4];
  const float* bv = (const float*)d_in[5];
  const float* Wk = (const float*)d_in[6];
  const float* bk = (const float*)d_in[7];
  const float* Wq = (const float*)d_in[8];
  const float* bq = (const float*)d_in[9];
  const float* Wo = (const float*)d_in[10];
  const float* bo = (const float*)d_in[11];
  float* out = (float*)d_out;

  const size_t buf = (size_t)NBATCH * L_SEQ * EMB;  // 8 Mi bf16 elements each
  unsigned short* qb = (unsigned short*)d_ws;
  unsigned short* kb = qb + buf;
  unsigned short* vb = kb + buf;
  unsigned short* ob = vb + buf;

  proj_qkv<<<dim3(EMB / 64, (NBATCH * L_SEQ) / 128, 3), 256, 0, stream>>>(
      query, keys, values, Wq, Wk, Wv, bq, bk, bv, qb, kb, vb);
  attn_flash<<<dim3(L_SEQ / 64, NBATCH * NHEAD), 128, 0, stream>>>(
      qb, kb, vb, mask, ob);
  out_proj<<<dim3(EMB / 64, (NBATCH * L_SEQ) / 128), 256, 0, stream>>>(
      ob, Wo, bo, out);
}
